// TemporalGNN_65781719106372
// MI455X (gfx1250) — matrix operand order repaired and checked by reading, not compiled, not run on hardware
//
#include <hip/hip_runtime.h>

typedef __attribute__((ext_vector_type(16))) _Float16 v16h;
typedef __attribute__((ext_vector_type(8)))  float    v8f;

#define IN_DIM   128
#define HID      256
#define OUT_DIM  128
#define B_SZ     2048
#define KFAN     16
#define N2_ROWS  34816   // B*K + B
#define E1_ROWS  32768   // B*K

// ---------------------------------------------------------------------------
// Weight pre-swizzle: B-fragment order for v_wmma_f32_16x16x32_f16.
// Fragment element p of lane (half*16 + n) holds B[k = c*32 + 16*half + p][ntile*16+n]
// Layer1: K=256 (W1n cols 0..127, W1h cols 128..255), N=256, KC=8, NT=16
// Layer2: K=512 (W2n cols 0..255, W2h cols 256..511), N=128, KC=16, NT=8
// ---------------------------------------------------------------------------
__global__ void prep_weights_kernel(const float* __restrict__ W1n, const float* __restrict__ W1h,
                                    const float* __restrict__ b1n, const float* __restrict__ b1h,
                                    const float* __restrict__ W2n, const float* __restrict__ W2h,
                                    const float* __restrict__ b2n, const float* __restrict__ b2h,
                                    _Float16* __restrict__ W1sw, _Float16* __restrict__ W2sw,
                                    float* __restrict__ bias1, float* __restrict__ bias2) {
    int gid = blockIdx.x * 256 + threadIdx.x;
    if (gid < 4096) {
        // layer 1: gid = ((ntile*8 + c)*32 + lane)
        int lane  = gid & 31;
        int c     = (gid >> 5) & 7;
        int ntile = gid >> 8;
        int n     = ntile * 16 + (lane & 15);
        int kbase = c * 32 + 16 * (lane >> 4);
        v16h frag;
        for (int p = 0; p < 16; ++p) {
            int k = kbase + p;
            float v = (k < 128) ? W1n[n * 128 + k] : W1h[n * 128 + (k - 128)];
            frag[p] = (_Float16)v;
        }
        *(v16h*)(W1sw + (size_t)gid * 16) = frag;
    } else if (gid < 8192) {
        int t     = gid - 4096;            // t = ((ntile*16 + c)*32 + lane)
        int lane  = t & 31;
        int c     = (t >> 5) & 15;
        int ntile = t >> 9;
        int n     = ntile * 16 + (lane & 15);
        int kbase = c * 32 + 16 * (lane >> 4);
        v16h frag;
        for (int p = 0; p < 16; ++p) {
            int k = kbase + p;
            float v = (k < 256) ? W2n[n * 256 + k] : W2h[n * 256 + (k - 256)];
            frag[p] = (_Float16)v;
        }
        *(v16h*)(W2sw + (size_t)t * 16) = frag;
    } else if (gid < 8448) {
        int n = gid - 8192;
        bias1[n] = b1n[n] + b1h[n];
        if (n < 128) bias2[n] = b2n[n] + b2h[n];
    }
}

// ---------------------------------------------------------------------------
// Layer-1 activation build: per wave, one frontier row i.
//   agg2[i] = sum_k x[nbr2[i*16+k]]            (128 f32)
//   hn1[i]  = x[nodes2[i]] * exp(-tk*(t2-nt2)) (128 f32)
// Row [agg2|hn1] (256 f32) staged in LDS, then stored f16 in A-fragment order:
//   element p of lane (16*half + m) holds A[m][k], k = 32c + 16*half + p
// ---------------------------------------------------------------------------
__global__ void prep_A1_kernel(const float* __restrict__ x, const int* __restrict__ nodes2,
                               const int* __restrict__ nbr2, const float* __restrict__ t2,
                               const float* __restrict__ n_times2,
                               const float* __restrict__ t_kernel,
                               _Float16* __restrict__ A1sw) {
    __shared__ float lds[8 * 256];
    const int wv   = threadIdx.x >> 5;
    const int lane = threadIdx.x & 31;
    const int row  = blockIdx.x * 8 + wv;

    float* ldsrow = lds + wv * 256;
    const float tk = t_kernel[0];

    // gather-sum 16 neighbor rows; lane covers 4 contiguous floats
    float ax = 0.f, ay = 0.f, az = 0.f, aw = 0.f;
    const int* nb = nbr2 + (size_t)row * KFAN;
    for (int k = 0; k < KFAN; ++k) {
        int nid = nb[k];
        float4 v = *((const float4*)(x + (size_t)nid * IN_DIM) + lane);
        ax += v.x; ay += v.y; az += v.z; aw += v.w;
    }
    int self = nodes2[row];
    float4 sv = *((const float4*)(x + (size_t)self * IN_DIM) + lane);
    float d = __expf(-tk * (t2[row] - n_times2[row]));

    float4 agg; agg.x = ax; agg.y = ay; agg.z = az; agg.w = aw;
    float4 hn;  hn.x = sv.x * d; hn.y = sv.y * d; hn.z = sv.z * d; hn.w = sv.w * d;
    ((float4*)ldsrow)[lane]         = agg;   // cols 0..127
    ((float4*)(ldsrow + 128))[lane] = hn;    // cols 128..255
    __syncthreads();

    // swizzle-store: 8 kchunks x 2 halves -> lanes 0..15
    if (lane < 16) {
        int c = lane >> 1, half = lane & 1;
        int mtile = row >> 4, m = row & 15;
        v16h frag;
        for (int p = 0; p < 16; ++p) {
            int k = c * 32 + 16 * half + p;
            frag[p] = (_Float16)ldsrow[k];
        }
        *(v16h*)(A1sw + ((size_t)(mtile * 8 + c) * 32 + 16 * half + m) * 16) = frag;
    }
}

// ---------------------------------------------------------------------------
// Layer-2 activation build: per wave, one seed b.
//   agg1[b] = sum_{k<16} h1[b*16+k]                  (256 f32)
//   hn2[b]  = h1[E1+b] * exp(-tk*(ts-nt1))           (256 f32)
// Row [agg1|hn2] (512 f32) -> swizzled f16, KC = 16.
// ---------------------------------------------------------------------------
__global__ void prep_A2_kernel(const float* __restrict__ h1, const float* __restrict__ ts,
                               const float* __restrict__ n_times1,
                               const float* __restrict__ t_kernel,
                               _Float16* __restrict__ A2sw) {
    __shared__ float lds[8 * 512];
    const int wv   = threadIdx.x >> 5;
    const int lane = threadIdx.x & 31;
    const int b    = blockIdx.x * 8 + wv;

    float* ldsrow = lds + wv * 512;
    const float tk = t_kernel[0];

    float a0x=0,a0y=0,a0z=0,a0w=0, a1x=0,a1y=0,a1z=0,a1w=0;
    for (int k = 0; k < KFAN; ++k) {
        const float4* src = (const float4*)(h1 + (size_t)(b * KFAN + k) * HID);
        float4 v0 = src[lane];
        float4 v1 = src[lane + 32];
        a0x += v0.x; a0y += v0.y; a0z += v0.z; a0w += v0.w;
        a1x += v1.x; a1y += v1.y; a1z += v1.z; a1w += v1.w;
    }
    float4 agg0; agg0.x=a0x; agg0.y=a0y; agg0.z=a0z; agg0.w=a0w;
    float4 agg1; agg1.x=a1x; agg1.y=a1y; agg1.z=a1z; agg1.w=a1w;
    ((float4*)ldsrow)[lane]      = agg0;   // cols 0..127
    ((float4*)ldsrow)[lane + 32] = agg1;   // cols 128..255

    float d = __expf(-tk * (ts[b] - n_times1[b]));
    const float4* selfrow = (const float4*)(h1 + (size_t)(E1_ROWS + b) * HID);
    float4 s0 = selfrow[lane], s1 = selfrow[lane + 32];
    s0.x *= d; s0.y *= d; s0.z *= d; s0.w *= d;
    s1.x *= d; s1.y *= d; s1.z *= d; s1.w *= d;
    ((float4*)(ldsrow + 256))[lane]      = s0;   // cols 256..383
    ((float4*)(ldsrow + 256))[lane + 32] = s1;   // cols 384..511
    __syncthreads();

    // 16 kchunks x 2 halves -> all 32 lanes
    int c = lane >> 1, half = lane & 1;
    int mtile = b >> 4, m = b & 15;
    v16h frag;
    for (int p = 0; p < 16; ++p) {
        int k = c * 32 + 16 * half + p;
        frag[p] = (_Float16)ldsrow[k];
    }
    *(v16h*)(A2sw + ((size_t)(mtile * 16 + c) * 32 + 16 * half + m) * 16) = frag;
}

// ---------------------------------------------------------------------------
// Register-blocked WMMA GEMM + bias + ReLU.
// Each wave computes an (MB*16) x (NB*16) output patch: per k-chunk it loads
// MB A-fragments + NB B-fragments (contiguous 32 B per lane) and issues MB*NB
// v_wmma_f32_16x16x32_f16 ops -> 8 WMMAs per 6 fragment loads at MB=2, NB=4.
// C layout: acc element r -> row (r + 8*(lane>>4)), col (lane&15).
// ---------------------------------------------------------------------------
template <int MB, int NB>
__global__ void gemm_relu_kernel(const _Float16* __restrict__ Asw,
                                 const _Float16* __restrict__ Bsw,
                                 const float* __restrict__ bias,
                                 float* __restrict__ outp,
                                 int ngroups, int kc, int ncols) {
    const int wv   = threadIdx.x >> 5;
    const int lane = threadIdx.x & 31;
    const int wid  = blockIdx.x * 8 + wv;
    const int mt0  = (wid / ngroups) * MB;   // first mtile of this wave
    const int ng   = wid % ngroups;          // group of NB ntiles

    const v16h* Ap = (const v16h*)Asw + (size_t)mt0 * kc * 32 + lane;
    const v16h* Bp = (const v16h*)Bsw + (size_t)(ng * NB) * kc * 32 + lane;

    v8f acc[MB][NB];
#pragma unroll
    for (int i = 0; i < MB; ++i)
#pragma unroll
        for (int j = 0; j < NB; ++j) acc[i][j] = (v8f){};

    for (int c = 0; c < kc; ++c) {
        v16h a[MB], b[NB];
#pragma unroll
        for (int i = 0; i < MB; ++i) a[i] = Ap[((size_t)i * kc + c) * 32];
#pragma unroll
        for (int j = 0; j < NB; ++j) b[j] = Bp[((size_t)j * kc + c) * 32];
#pragma unroll
        for (int i = 0; i < MB; ++i)
#pragma unroll
            for (int j = 0; j < NB; ++j)
                acc[i][j] = __builtin_amdgcn_wmma_f32_16x16x32_f16(
                    false, a[i], false, b[j], (short)0, acc[i][j], false, false);
    }

    const int half = lane >> 4;
#pragma unroll
    for (int i = 0; i < MB; ++i) {
        const int row0 = (mt0 + i) * 16 + 8 * half;
#pragma unroll
        for (int j = 0; j < NB; ++j) {
            const int col = (ng * NB + j) * 16 + (lane & 15);
            const float bc = bias[col];
#pragma unroll
            for (int r = 0; r < 8; ++r) {
                float v = acc[i][j][r] + bc;
                outp[(size_t)(row0 + r) * ncols + col] = fmaxf(v, 0.f);
            }
        }
    }
}

// ---------------------------------------------------------------------------
extern "C" void kernel_launch(void* const* d_in, const int* in_sizes, int n_in,
                              void* d_out, int out_size, void* d_ws, size_t ws_size,
                              hipStream_t stream) {
    const float* x        = (const float*)d_in[0];
    const float* ts       = (const float*)d_in[1];
    const float* t2       = (const float*)d_in[2];
    const float* n_times1 = (const float*)d_in[3];
    const float* n_times2 = (const float*)d_in[4];
    const float* W1n      = (const float*)d_in[5];
    const float* b1n      = (const float*)d_in[6];
    const float* W1h      = (const float*)d_in[7];
    const float* b1h      = (const float*)d_in[8];
    const float* W2n      = (const float*)d_in[9];
    const float* b2n      = (const float*)d_in[10];
    const float* W2h      = (const float*)d_in[11];
    const float* b2h      = (const float*)d_in[12];
    const float* t_kernel = (const float*)d_in[13];
    const int*   nodes2   = (const int*)d_in[14];
    const int*   nbr2     = (const int*)d_in[15];
    // d_in[16] = seg2, d_in[17] = seg1 (implicit in contiguous grouping of K)

    char* ws = (char*)d_ws;
    size_t off = 0;
    auto alloc = [&](size_t bytes) -> void* {
        void* p = ws + off;
        off = (off + bytes + 255) & ~(size_t)255;
        return p;
    };
    _Float16* W1sw  = (_Float16*)alloc((size_t)256 * 256 * 2);   // 128 KB
    _Float16* W2sw  = (_Float16*)alloc((size_t)512 * 128 * 2);   // 128 KB
    float*    bias1 = (float*)alloc(256 * 4);
    float*    bias2 = (float*)alloc(128 * 4);
    _Float16* A1sw  = (_Float16*)alloc((size_t)N2_ROWS * 256 * 2); // ~17.8 MB
    _Float16* A2sw  = (_Float16*)alloc((size_t)B_SZ * 512 * 2);    // 2 MB
    float*    h1    = (float*)alloc((size_t)N2_ROWS * HID * 4);    // ~35.7 MB

    // 1) fold + swizzle weights, sum biases
    prep_weights_kernel<<<33, 256, 0, stream>>>(W1n, W1h, b1n, b1h,
                                                W2n, W2h, b2n, b2h,
                                                W1sw, W2sw, bias1, bias2);

    // 2) hop-2 gather + decay -> swizzled [agg2|hn1] f16, M=N2, K=256
    prep_A1_kernel<<<N2_ROWS / 8, 256, 0, stream>>>(x, nodes2, nbr2, t2, n_times2,
                                                    t_kernel, A1sw);

    // 3) h1 = relu(A1 @ W1c^T + bias1): M=34816 -> 2176 mtiles (MB=2 -> 1088 mgroups),
    //    N=256 -> 16 ntiles (NB=4 -> 4 ngroups), KC=8. 4352 waves -> 544 blocks.
    gemm_relu_kernel<2, 4><<<544, 256, 0, stream>>>(A1sw, W1sw, bias1, h1,
                                                    /*ngroups=*/4, /*kc=*/8,
                                                    /*ncols=*/HID);

    // 4) segment-sum + decay -> swizzled [agg1|hn2] f16, M=2048, K=512
    prep_A2_kernel<<<B_SZ / 8, 256, 0, stream>>>(h1, ts, n_times1, t_kernel, A2sw);

    // 5) out = relu(A2 @ W2c^T + bias2): M=2048 -> 128 mtiles (MB=2 -> 64 mgroups),
    //    N=128 -> 8 ntiles (NB=4 -> 2 ngroups), KC=16. 128 waves -> 16 blocks.
    gemm_relu_kernel<2, 4><<<16, 256, 0, stream>>>(A2sw, W2sw, bias2, (float*)d_out,
                                                   /*ngroups=*/2, /*kc=*/16,
                                                   /*ncols=*/OUT_DIM);
}
